// STDCNN_78537771975094
// MI455X (gfx1250) — matrix operand order repaired and checked by reading, not compiled, not run on hardware
//
#include <hip/hip_runtime.h>
#include <hip/hip_fp16.h>

// ---------------------------------------------------------------------------
// STDCNN pipeline for MI455X (gfx1250, wave32).
// Convs are BN-folded + reassociated (channel-mix-first for branch 1, fully
// fused effective weights for branch 2). Transformer dense layers use
// v_wmma_f32_16x16x32_f16 (552 row-tiles of 16x16, K padded to 32) with
// branchless vectorized (b128) operand loaders. Attention (head_dim=2) is a
// fused flash-style VALU kernel with b128 LDS staging.
// ---------------------------------------------------------------------------

typedef __attribute__((ext_vector_type(16))) _Float16 v16h;
typedef __attribute__((ext_vector_type(8)))  float    v8f;

#define B_    128
#define NCH   22
#define T_    1000
#define F1    8
#define F2    16
#define L1    31
#define L2    38
#define LSEQ  69
#define EMB   16
#define NROWS (B_ * LSEQ)        // 8832
#define NTILE (NROWS / 16)       // 552

// ---------------------------------------------------------------------------
// K0: fold all BatchNorms into effective conv weights / biases.
// ---------------------------------------------------------------------------
__global__ void k_prep(const float* __restrict__ w_spec1, const float* __restrict__ b_spec1,
                       const float* __restrict__ w_spec2, const float* __restrict__ b_spec2,
                       const float* __restrict__ g_bn1,  const float* __restrict__ b_bn1,
                       const float* __restrict__ g_bn2,  const float* __restrict__ b_bn2,
                       const float* __restrict__ w_sp1a, const float* __restrict__ g_bn3,
                       const float* __restrict__ b_bn3,  const float* __restrict__ w_sp1b,
                       const float* __restrict__ b_sp1b, const float* __restrict__ g_bn4,
                       const float* __restrict__ b_bn4,  const float* __restrict__ w_sp2,
                       const float* __restrict__ b_sp2,  const float* __restrict__ g_bn5,
                       const float* __restrict__ b_bn5,
                       float* __restrict__ w1eff, float* __restrict__ cst1,
                       float* __restrict__ w2eff, float* __restrict__ cst2,
                       float* __restrict__ w1beff, float* __restrict__ b1beff) {
  const float rs = rsqrtf(1.0f + 1e-5f);
  for (int i = threadIdx.x; i < F2 * 125; i += blockDim.x) {
    int f2 = i / 125, k = i % 125, g = f2 >> 1;
    w1eff[i] = (g_bn3[f2] * rs) * (g_bn1[g] * rs) * w_spec1[g * 125 + k];
  }
  for (int f2 = threadIdx.x; f2 < F2; f2 += blockDim.x) {
    int g = f2 >> 1;
    float s1 = g_bn1[g] * rs;
    float sh1 = b_spec1[g] * s1 + b_bn1[g];
    float sumwa = 0.f;
    for (int c = 0; c < NCH; ++c) sumwa += w_sp1a[f2 * NCH + c];
    cst1[f2] = (g_bn3[f2] * rs) * sh1 * sumwa + b_bn3[f2];
  }
  for (int i = threadIdx.x; i < F2 * NCH * 30; i += blockDim.x) {
    int f2 = i / (NCH * 30), r = i % (NCH * 30), c = r / 30, k = r % 30;
    float acc = 0.f;
    for (int f1 = 0; f1 < F1; ++f1)
      acc += (g_bn2[f1] * rs) * w_sp2[f2 * F1 * NCH + f1 * NCH + c] * w_spec2[f1 * 30 + k];
    w2eff[i] = (g_bn5[f2] * rs) * acc;
  }
  for (int f2 = threadIdx.x; f2 < F2; f2 += blockDim.x) {
    float acc = 0.f;
    for (int f1 = 0; f1 < F1; ++f1) {
      float s2 = g_bn2[f1] * rs;
      float sh2 = b_spec2[f1] * s2 + b_bn2[f1];
      float sw = 0.f;
      for (int c = 0; c < NCH; ++c) sw += w_sp2[f2 * F1 * NCH + f1 * NCH + c];
      acc += sw * sh2;
    }
    cst2[f2] = (g_bn5[f2] * rs) * (acc + b_sp2[f2]) + b_bn5[f2];
  }
  for (int i = threadIdx.x; i < F2 * F2; i += blockDim.x) {
    int f2 = i / F2;
    w1beff[i] = (g_bn4[f2] * rs) * w_sp1b[i];
  }
  for (int f2 = threadIdx.x; f2 < F2; f2 += blockDim.x)
    b1beff[f2] = (g_bn4[f2] * rs) * b_sp1b[f2] + b_bn4[f2];
}

// ---------------------------------------------------------------------------
// K1: channel mix ymix[b,f2,t] = sum_c wa[f2,c] * x[b,c,t]
// ---------------------------------------------------------------------------
__global__ void k_ymix(const float* __restrict__ x, const float* __restrict__ w_sp1a,
                       float* __restrict__ ymix) {
  int bf = blockIdx.x;
  int b = bf >> 4, f2 = bf & 15;
  const float* xb = x + (size_t)b * NCH * T_;
  float w[NCH];
#pragma unroll
  for (int c = 0; c < NCH; ++c) w[c] = w_sp1a[f2 * NCH + c];
  for (int t = threadIdx.x; t < T_; t += blockDim.x) {
    float acc = 0.f;
#pragma unroll
    for (int c = 0; c < NCH; ++c) acc = fmaf(w[c], xb[c * T_ + t], acc);
    ymix[(size_t)bf * T_ + t] = acc;
  }
}

// ---------------------------------------------------------------------------
// K2: branch-1 temporal conv (k=125, same pad 62) + BN3 + ELU.
// Fast path (fully in range) has no per-tap predication.
// ---------------------------------------------------------------------------
__global__ void k_s1a(const float* __restrict__ ymix, const float* __restrict__ w1eff,
                      const float* __restrict__ cst1, float* __restrict__ s1a) {
  int bf = blockIdx.x, f2 = bf & 15;
  const float* y = ymix + (size_t)bf * T_;
  const float* w = w1eff + f2 * 125;
  float c0 = cst1[f2];
  for (int t = threadIdx.x; t < T_; t += blockDim.x) {
    float acc = c0;
    int lo = t - 62;
    if (lo >= 0 && lo + 125 <= T_) {
      const float* yy = y + lo;
#pragma unroll 5
      for (int k = 0; k < 125; ++k) acc = fmaf(w[k], yy[k], acc);
    } else {
      int k0 = (lo < 0) ? -lo : 0;
      int k1 = (lo + 125 > T_) ? (T_ - lo) : 125;
      for (int k = k0; k < k1; ++k) acc = fmaf(w[k], y[lo + k], acc);
    }
    s1a[(size_t)bf * T_ + t] = (acc > 0.f) ? acc : (__expf(acc) - 1.f);
  }
}

// ---------------------------------------------------------------------------
// K3: 1x1 channel conv + BN4 + ELU + avgpool(32,32) -> p1 (B,16,31)
// ---------------------------------------------------------------------------
__global__ void k_s1b_pool(const float* __restrict__ s1a, const float* __restrict__ w1beff,
                           const float* __restrict__ b1beff, float* __restrict__ p1) {
  int gid = blockIdx.x * blockDim.x + threadIdx.x;
  if (gid >= B_ * F2 * L1) return;
  int j = gid % L1, f2 = (gid / L1) % F2, b = gid / (L1 * F2);
  const float* wrow = w1beff + f2 * F2;
  float bias = b1beff[f2];
  float acc = 0.f;
  for (int i = 0; i < 32; ++i) {
    int t = 32 * j + i;
    float v = bias;
#pragma unroll
    for (int g = 0; g < F2; ++g)
      v = fmaf(wrow[g], s1a[(size_t)(b * F2 + g) * T_ + t], v);
    v = (v > 0.f) ? v : (__expf(v) - 1.f);
    acc += v;
  }
  p1[gid] = acc * (1.f / 32.f);
}

// ---------------------------------------------------------------------------
// K4a: branch-2 fused conv (k=30, 22 chans, pad 14/15) + BN5, squared
// ---------------------------------------------------------------------------
__global__ void k_s2(const float* __restrict__ x, const float* __restrict__ w2eff,
                     const float* __restrict__ cst2, float* __restrict__ s2sq) {
  int bf = blockIdx.x;
  int b = bf >> 4, f2 = bf & 15;
  const float* xb = x + (size_t)b * NCH * T_;
  const float* w = w2eff + f2 * NCH * 30;
  float c0 = cst2[f2];
  for (int t = threadIdx.x; t < T_; t += blockDim.x) {
    float acc = c0;
    int lo = t - 14;
    if (lo >= 0 && lo + 30 <= T_) {
      for (int c = 0; c < NCH; ++c) {
        const float* xr = xb + c * T_ + lo;
        const float* wr = w + c * 30;
#pragma unroll
        for (int k = 0; k < 30; ++k) acc = fmaf(wr[k], xr[k], acc);
      }
    } else {
      int k0 = (lo < 0) ? -lo : 0;
      int k1 = (lo + 30 > T_) ? (T_ - lo) : 30;
      for (int c = 0; c < NCH; ++c) {
        const float* xr = xb + c * T_;
        const float* wr = w + c * 30;
        for (int k = k0; k < k1; ++k) acc = fmaf(wr[k], xr[lo + k], acc);
      }
    }
    s2sq[(size_t)bf * T_ + t] = acc * acc;
  }
}

// K4b: avgpool(75,25) + log(clip) -> p2 (B,16,38)
__global__ void k_pool2(const float* __restrict__ s2sq, float* __restrict__ p2) {
  int gid = blockIdx.x * blockDim.x + threadIdx.x;
  if (gid >= B_ * F2 * L2) return;
  int j = gid % L2, bf = gid / L2;
  const float* s = s2sq + (size_t)bf * T_ + 25 * j;
  float acc = 0.f;
  for (int i = 0; i < 75; ++i) acc += s[i];
  acc *= (1.f / 75.f);
  p2[gid] = logf(fmaxf(acc, 1e-6f));
}

// ---------------------------------------------------------------------------
// K5: linear SSM scan over L=69; block = 8 batches (128 thr = 8b x 16f)
// ---------------------------------------------------------------------------
__global__ void k_ssm(const float* __restrict__ p1, const float* __restrict__ p2,
                      const float* __restrict__ A, const float* __restrict__ Bm,
                      const float* __restrict__ Cm, float* __restrict__ h) {
  __shared__ float st[8][F2];
  __shared__ float nst[8][F2];
  int bl = threadIdx.x >> 4;
  int f  = threadIdx.x & 15;
  int b  = blockIdx.x * 8 + bl;
  st[bl][f] = 0.f;
  __syncthreads();
  for (int l = 0; l < LSEQ; ++l) {
    float acc = 0.f;
#pragma unroll
    for (int g = 0; g < F2; ++g) {
      float xg = (l < L1) ? p1[(b * F2 + g) * L1 + l]
                          : p2[(b * F2 + g) * L2 + (l - L1)];
      acc = fmaf(xg, Bm[f * F2 + g], acc);
    }
#pragma unroll
    for (int g = 0; g < F2; ++g) acc = fmaf(st[bl][g], A[g * F2 + f], acc);
    nst[bl][f] = acc;
    __syncthreads();
    float o = 0.f;
#pragma unroll
    for (int g = 0; g < F2; ++g) o = fmaf(nst[bl][g], Cm[f * F2 + g], o);
    h[((size_t)b * LSEQ + l) * EMB + f] = o;
    st[bl][f] = nst[bl][f];
    __syncthreads();
  }
}

// ---------------------------------------------------------------------------
// Branchless WMMA operand loaders (wave32). ISA 7.12.2 layouts:
//  A (16x32 f16): lane m=lane%16 holds row M=m; elems 0..7 -> K = 8*(lane/16)+e,
//                 elems 8..15 -> K = 16 + 8*(lane/16) + (e-8)
//  B (32x16 f16): lane n=lane%16 holds col N=n; elem e -> same K as A's elem e
//  C/D (16x16 f32): elem r -> M = r + 8*(lane/16), N = lane%16
// All loads are unconditional b128 from valid addresses; zero-padding is done
// on values (no EXEC divergence around WMMA).
// ---------------------------------------------------------------------------
__device__ __forceinline__ v16h load_A_k16(const float* __restrict__ base, int ld) {
  int lane = threadIdx.x & 31;
  int m = lane & 15;
  int hi = lane >> 4;
  const float4* p = reinterpret_cast<const float4*>(base + m * ld + hi * 8);
  float4 x0 = p[0], x1 = p[1];
  v16h a = {};  // elems 8..15 (K=16..31) stay zero
  a[0] = (_Float16)x0.x; a[1] = (_Float16)x0.y; a[2] = (_Float16)x0.z; a[3] = (_Float16)x0.w;
  a[4] = (_Float16)x1.x; a[5] = (_Float16)x1.y; a[6] = (_Float16)x1.z; a[7] = (_Float16)x1.w;
  return a;
}

__device__ __forceinline__ v16h load_A_full(const float* base, int ld, int kbase) {
  int lane = threadIdx.x & 31;
  int m = lane & 15;
  int hi = lane >> 4;
  const float* r = base + m * ld + kbase + hi * 8;
  const float4* p0 = reinterpret_cast<const float4*>(r);
  const float4* p1 = reinterpret_cast<const float4*>(r + 16);
  float4 x0 = p0[0], x1 = p0[1], y0 = p1[0], y1 = p1[1];
  v16h a;
  a[0]  = (_Float16)x0.x; a[1]  = (_Float16)x0.y; a[2]  = (_Float16)x0.z; a[3]  = (_Float16)x0.w;
  a[4]  = (_Float16)x1.x; a[5]  = (_Float16)x1.y; a[6]  = (_Float16)x1.z; a[7]  = (_Float16)x1.w;
  a[8]  = (_Float16)y0.x; a[9]  = (_Float16)y0.y; a[10] = (_Float16)y0.z; a[11] = (_Float16)y0.w;
  a[12] = (_Float16)y1.x; a[13] = (_Float16)y1.y; a[14] = (_Float16)y1.z; a[15] = (_Float16)y1.w;
  return a;
}

__device__ __forceinline__ v16h load_B_k16(const float* __restrict__ w, int ldk, int nbase) {
  int lane = threadIdx.x & 31;
  int n = lane & 15;
  const float4* p = reinterpret_cast<const float4*>(w + (nbase + n) * ldk + ((lane >> 4) << 3));
  float4 x0 = p[0], x1 = p[1];
  v16h b = {};  // elems 0..7 -> K = 8*(lane/16)+e; elems 8..15 (K=16..31) stay zero
  b[0]  = (_Float16)x0.x; b[1]  = (_Float16)x0.y;
  b[2]  = (_Float16)x0.z; b[3]  = (_Float16)x0.w;
  b[4]  = (_Float16)x1.x; b[5]  = (_Float16)x1.y;
  b[6]  = (_Float16)x1.z; b[7]  = (_Float16)x1.w;
  return b;
}

__device__ __forceinline__ v16h load_B_full(const float* __restrict__ w, int ldk,
                                            int nbase, int kbase) {
  int lane = threadIdx.x & 31;
  int n = lane & 15;
  int klo = (lane >> 4) << 3;
  const float4* p = reinterpret_cast<const float4*>(w + (nbase + n) * ldk + kbase + klo);
  float4 x0 = p[0], x1 = p[1], x2 = p[4], x3 = p[5];
  v16h b;
  b[0]  = (_Float16)x0.x; b[1]  = (_Float16)x0.y; b[2]  = (_Float16)x0.z; b[3]  = (_Float16)x0.w;
  b[4]  = (_Float16)x1.x; b[5]  = (_Float16)x1.y; b[6]  = (_Float16)x1.z; b[7]  = (_Float16)x1.w;
  b[8]  = (_Float16)x2.x; b[9]  = (_Float16)x2.y; b[10] = (_Float16)x2.z; b[11] = (_Float16)x2.w;
  b[12] = (_Float16)x3.x; b[13] = (_Float16)x3.y; b[14] = (_Float16)x3.z; b[15] = (_Float16)x3.w;
  return b;
}

// ---------------------------------------------------------------------------
// T1: QKV projection, one wave per 16-row tile; 3 WMMAs (N-tiles of 48)
// ---------------------------------------------------------------------------
__global__ void k_qkv(const float* __restrict__ h, const float* __restrict__ qw,
                      const float* __restrict__ qb, float* __restrict__ qkv) {
  int rt = blockIdx.x;
  const float* hrow = h + (size_t)rt * 16 * EMB;
  v16h a = load_A_k16(hrow, EMB);
  int lane = threadIdx.x & 31;
  int Nn = lane & 15;
  int Mg = (lane >> 4) << 3;
#pragma unroll
  for (int nb = 0; nb < 3; ++nb) {
    v16h b = load_B_k16(qw, EMB, nb * 16);
    v8f c = {};
    c = __builtin_amdgcn_wmma_f32_16x16x32_f16(false, a, false, b, (short)0, c,
                                               false, false);
    float bias = qb[nb * 16 + Nn];
#pragma unroll
    for (int r = 0; r < 8; ++r) {
      int row = rt * 16 + Mg + r;
      qkv[(size_t)row * 48 + nb * 16 + Nn] = c[r] + bias;
    }
  }
}

// ---------------------------------------------------------------------------
// T2: fused attention (8 heads, hd=2, S=128) + proj + residual + LayerNorm.
// Flash-style online softmax; K/V staged in LDS via b128.
// ---------------------------------------------------------------------------
__global__ void k_attn(const float* __restrict__ qkv, const float* __restrict__ hin,
                       const float* __restrict__ pw, const float* __restrict__ pb,
                       const float* __restrict__ g1, const float* __restrict__ b1,
                       float* __restrict__ hout) {
  __shared__ float ks[128][EMB];
  __shared__ float vs[128][EMB];
  int n = blockIdx.x;
  int s = threadIdx.x;
  int row = s * LSEQ + n;
  const float4* q4 = reinterpret_cast<const float4*>(qkv + (size_t)row * 48);
  float4 qv[4];
  float4* ks4 = reinterpret_cast<float4*>(&ks[s][0]);
  float4* vs4 = reinterpret_cast<float4*>(&vs[s][0]);
#pragma unroll
  for (int i = 0; i < 4; ++i) {
    qv[i]  = q4[i];
    ks4[i] = q4[4 + i];
    vs4[i] = q4[8 + i];
  }
  float q[EMB];
#pragma unroll
  for (int i = 0; i < 4; ++i) {
    q[4 * i] = qv[i].x; q[4 * i + 1] = qv[i].y; q[4 * i + 2] = qv[i].z; q[4 * i + 3] = qv[i].w;
  }
  __syncthreads();
  const float scale = 0.70710678f;  // 1/sqrt(head_dim=2)
  float m[8], l[8], a0[8], a1[8];
#pragma unroll
  for (int hh = 0; hh < 8; ++hh) { m[hh] = -1e30f; l[hh] = 0.f; a0[hh] = 0.f; a1[hh] = 0.f; }
  for (int t = 0; t < 128; ++t) {
    const float4* kr = reinterpret_cast<const float4*>(&ks[t][0]);
    const float4* vr = reinterpret_cast<const float4*>(&vs[t][0]);
    float kk[EMB], vv[EMB];
#pragma unroll
    for (int i = 0; i < 4; ++i) {
      float4 kq = kr[i], vq = vr[i];
      kk[4 * i] = kq.x; kk[4 * i + 1] = kq.y; kk[4 * i + 2] = kq.z; kk[4 * i + 3] = kq.w;
      vv[4 * i] = vq.x; vv[4 * i + 1] = vq.y; vv[4 * i + 2] = vq.z; vv[4 * i + 3] = vq.w;
    }
#pragma unroll
    for (int hh = 0; hh < 8; ++hh) {
      float sc = (q[2 * hh] * kk[2 * hh] + q[2 * hh + 1] * kk[2 * hh + 1]) * scale;
      float nm = fmaxf(m[hh], sc);
      float corr = __expf(m[hh] - nm);
      float p = __expf(sc - nm);
      l[hh] = l[hh] * corr + p;
      a0[hh] = a0[hh] * corr + p * vv[2 * hh];
      a1[hh] = a1[hh] * corr + p * vv[2 * hh + 1];
      m[hh] = nm;
    }
  }
  float o[EMB];
#pragma unroll
  for (int hh = 0; hh < 8; ++hh) {
    float inv = 1.f / l[hh];
    o[2 * hh] = a0[hh] * inv;
    o[2 * hh + 1] = a1[hh] * inv;
  }
  float r[EMB];
  float mean = 0.f;
#pragma unroll
  for (int j = 0; j < EMB; ++j) {
    float acc = pb[j];
#pragma unroll
    for (int e = 0; e < EMB; ++e) acc = fmaf(pw[j * EMB + e], o[e], acc);
    r[j] = hin[(size_t)row * EMB + j] + acc;
    mean += r[j];
  }
  mean *= (1.f / 16.f);
  float var = 0.f;
#pragma unroll
  for (int j = 0; j < EMB; ++j) { float d = r[j] - mean; var += d * d; }
  var *= (1.f / 16.f);
  float inv = rsqrtf(var + 1e-5f);
#pragma unroll
  for (int j = 0; j < EMB; ++j)
    hout[(size_t)row * EMB + j] = (r[j] - mean) * inv * g1[j] + b1[j];
}

// ---------------------------------------------------------------------------
// T3: feed-forward 16->64(relu)->16 with WMMA, + residual + LayerNorm.
// GEMM1: 4 WMMAs (N=64); D->A transpose via LDS; GEMM2: two chained k32 WMMAs.
// ---------------------------------------------------------------------------
__global__ void k_ffn(const float* __restrict__ h, const float* __restrict__ w1,
                      const float* __restrict__ fb1, const float* __restrict__ w2,
                      const float* __restrict__ fb2, const float* __restrict__ g2,
                      const float* __restrict__ b2v, float* __restrict__ hout) {
  __shared__ float u[16][68];
  int rt = blockIdx.x;
  int lane = threadIdx.x & 31;
  int Nn = lane & 15;
  int Mg = (lane >> 4) << 3;
  const float* hrow = h + (size_t)rt * 16 * EMB;
  v16h a = load_A_k16(hrow, EMB);
#pragma unroll
  for (int nb = 0; nb < 4; ++nb) {
    v16h b = load_B_k16(w1, EMB, nb * 16);
    v8f c = {};
    c = __builtin_amdgcn_wmma_f32_16x16x32_f16(false, a, false, b, (short)0, c,
                                               false, false);
    float bias = fb1[nb * 16 + Nn];
#pragma unroll
    for (int r = 0; r < 8; ++r)
      u[Mg + r][nb * 16 + Nn] = fmaxf(c[r] + bias, 0.f);
  }
  __syncthreads();
  v16h a0 = load_A_full(&u[0][0], 68, 0);
  v16h a1 = load_A_full(&u[0][0], 68, 32);
  v16h b0 = load_B_full(w2, 64, 0, 0);
  v16h b1 = load_B_full(w2, 64, 0, 32);
  v8f c = {};
  c = __builtin_amdgcn_wmma_f32_16x16x32_f16(false, a0, false, b0, (short)0, c,
                                             false, false);
  c = __builtin_amdgcn_wmma_f32_16x16x32_f16(false, a1, false, b1, (short)0, c,
                                             false, false);
  __syncthreads();
#pragma unroll
  for (int r = 0; r < 8; ++r) {
    int M = Mg + r;
    u[M][Nn] = c[r] + fb2[Nn] + h[((size_t)rt * 16 + M) * EMB + Nn];
  }
  __syncthreads();
  if (lane < 16) {
    int M = lane;
    float mean = 0.f;
#pragma unroll
    for (int j = 0; j < EMB; ++j) mean += u[M][j];
    mean *= (1.f / 16.f);
    float var = 0.f;
#pragma unroll
    for (int j = 0; j < EMB; ++j) { float d = u[M][j] - mean; var += d * d; }
    var *= (1.f / 16.f);
    float inv = rsqrtf(var + 1e-5f);
#pragma unroll
    for (int j = 0; j < EMB; ++j)
      hout[((size_t)rt * 16 + M) * EMB + j] = (u[M][j] - mean) * inv * g2[j] + b2v[j];
  }
}

// ---------------------------------------------------------------------------
// Final transpose: (B, L, 16) -> (B, 16, 1, L)
// ---------------------------------------------------------------------------
__global__ void k_out(const float* __restrict__ h, float* __restrict__ out) {
  int gid = blockIdx.x * blockDim.x + threadIdx.x;
  if (gid >= B_ * F2 * LSEQ) return;
  int l = gid % LSEQ, f = (gid / LSEQ) % F2, b = gid / (LSEQ * F2);
  out[gid] = h[((size_t)b * LSEQ + l) * EMB + f];
}

// ---------------------------------------------------------------------------
extern "C" void kernel_launch(void* const* d_in, const int* in_sizes, int n_in,
                              void* d_out, int out_size, void* d_ws, size_t ws_size,
                              hipStream_t stream) {
  (void)in_sizes; (void)n_in; (void)out_size; (void)ws_size;
  const float* x       = (const float*)d_in[0];
  const float* w_spec1 = (const float*)d_in[1];
  const float* b_spec1 = (const float*)d_in[2];
  const float* w_spec2 = (const float*)d_in[3];
  const float* b_spec2 = (const float*)d_in[4];
  const float* g_bn1   = (const float*)d_in[5];
  const float* b_bn1   = (const float*)d_in[6];
  const float* g_bn2   = (const float*)d_in[7];
  const float* b_bn2   = (const float*)d_in[8];
  const float* w_sp1a  = (const float*)d_in[9];
  const float* g_bn3   = (const float*)d_in[10];
  const float* b_bn3   = (const float*)d_in[11];
  const float* w_sp1b  = (const float*)d_in[12];
  const float* b_sp1b  = (const float*)d_in[13];
  const float* g_bn4   = (const float*)d_in[14];
  const float* b_bn4   = (const float*)d_in[15];
  const float* w_sp2   = (const float*)d_in[16];
  const float* b_sp2   = (const float*)d_in[17];
  const float* g_bn5   = (const float*)d_in[18];
  const float* b_bn5   = (const float*)d_in[19];
  const float* Amat    = (const float*)d_in[20];
  const float* Bm      = (const float*)d_in[21];
  const float* Cm      = (const float*)d_in[22];
  const float* qkv_w   = (const float*)d_in[23];
  const float* qkv_b   = (const float*)d_in[24];
  const float* proj_w  = (const float*)d_in[25];
  const float* proj_b  = (const float*)d_in[26];
  const float* ff1_w   = (const float*)d_in[27];
  const float* ff1_b   = (const float*)d_in[28];
  const float* ff2_w   = (const float*)d_in[29];
  const float* ff2_b   = (const float*)d_in[30];
  const float* ln1_g   = (const float*)d_in[31];
  const float* ln1_b   = (const float*)d_in[32];
  const float* ln2_g   = (const float*)d_in[33];
  const float* ln2_b   = (const float*)d_in[34];

  float* ws = (float*)d_ws;
  float* w1eff  = ws;                       //  2000
  float* cst1   = w1eff + 2000;             //    16
  float* w2eff  = cst1 + 16;                // 10560
  float* cst2   = w2eff + 10560;            //    16
  float* w1beff = cst2 + 16;                //   256
  float* b1beff = w1beff + 256;             //    16
  float* ymix   = ws + 12864;               // 128*16*1000
  float* s1a    = ymix + 2048000;           // 128*16*1000 (reused as s2sq)
  float* p1     = s1a + 2048000;            // 128*16*31
  float* p2     = p1 + (B_ * F2 * L1);      // 128*16*38
  float* hA     = p2 + (B_ * F2 * L2);      // 8832*16
  float* hB     = hA + NROWS * EMB;         // 8832*16
  float* qkv    = hB + NROWS * EMB;         // 8832*48
  float* s2sq   = s1a;                      // buffer reuse (sequential)

  k_prep<<<1, 256, 0, stream>>>(w_spec1, b_spec1, w_spec2, b_spec2,
                                g_bn1, b_bn1, g_bn2, b_bn2,
                                w_sp1a, g_bn3, b_bn3, w_sp1b, b_sp1b,
                                g_bn4, b_bn4, w_sp2, b_sp2, g_bn5, b_bn5,
                                w1eff, cst1, w2eff, cst2, w1beff, b1beff);

  k_ymix<<<B_ * F2, 256, 0, stream>>>(x, w_sp1a, ymix);
  k_s1a <<<B_ * F2, 256, 0, stream>>>(ymix, w1eff, cst1, s1a);
  k_s1b_pool<<<(B_ * F2 * L1 + 255) / 256, 256, 0, stream>>>(s1a, w1beff, b1beff, p1);
  k_s2  <<<B_ * F2, 256, 0, stream>>>(x, w2eff, cst2, s2sq);
  k_pool2<<<(B_ * F2 * L2 + 255) / 256, 256, 0, stream>>>(s2sq, p2);
  k_ssm <<<B_ / 8, 128, 0, stream>>>(p1, p2, Amat, Bm, Cm, hA);

  for (int l = 0; l < 2; ++l) {
    const float* qw = qkv_w + l * 48 * EMB;
    const float* qb = qkv_b + l * 48;
    const float* pw = proj_w + l * EMB * EMB;
    const float* pb = proj_b + l * EMB;
    const float* f1w = ff1_w + l * 64 * EMB;
    const float* f1b = ff1_b + l * 64;
    const float* f2w = ff2_w + l * EMB * 64;
    const float* f2b = ff2_b + l * EMB;
    k_qkv <<<NTILE, 32, 0, stream>>>(hA, qw, qb, qkv);
    k_attn<<<LSEQ, 128, 0, stream>>>(qkv, hA, pw, pb, ln1_g + l * EMB, ln1_b + l * EMB, hB);
    k_ffn <<<NTILE, 32, 0, stream>>>(hB, f1w, f1b, f2w, f2b, ln2_g + l * EMB, ln2_b + l * EMB, hA);
  }

  k_out<<<(B_ * F2 * LSEQ + 255) / 256, 256, 0, stream>>>(hA, (float*)d_out);
}
